// CustomMultiHeadAttention_26422638805283
// MI455X (gfx1250) — matrix operand order repaired and checked by reading, not compiled, not run on hardware
//
#include <hip/hip_runtime.h>

#define EMBED 2048
#define TT    2048
#define BB    2
#define HEADS 16
#define DH    128
#define NTOK  (BB*TT)   // 4096
#define LDSP  40        // padded LDS row pitch (elements) for 32-wide k-slices

typedef __attribute__((ext_vector_type(16))) __bf16 v16bf;
typedef __attribute__((ext_vector_type(8)))  float  v8f;

union Frag16 {
  v16bf v;
  uint4 q[2];
  unsigned short us[16];
};

__device__ __forceinline__ unsigned short f2bf(float f) {
  unsigned u = __float_as_uint(f);
  u += 0x7FFFu + ((u >> 16) & 1u);      // round to nearest even
  return (unsigned short)(u >> 16);
}

// ---- global-memory fragment loaders (attention kernel) ----
// A-matrix 16x32 bf16 fragment: lane r(0..15) half c holds K = c*16 .. c*16+15;
// src row-major, K contiguous.
__device__ __forceinline__ v16bf load_afrag(const unsigned short* base, int ld, int row0, int k0) {
  const int lane = threadIdx.x & 31;
  const int r = lane & 15, c = lane >> 4;
  const unsigned short* p = base + (size_t)(row0 + r) * ld + k0 + c * 16;
  Frag16 f;
  f.q[0] = *(const uint4*)(p);
  f.q[1] = *(const uint4*)(p + 8);
  return f.v;
}

// B-matrix 32x16 bf16 fragment: lane holds column n = lane&15, K = c*16 .. c*16+15.
__device__ __forceinline__ v16bf load_bfrag(const unsigned short* base, int ld, int n0, int k0) {
  const int lane = threadIdx.x & 31;
  const int n = lane & 15, c = lane >> 4;
  const unsigned short* p = base + (size_t)(n0 + n) * ld + k0 + c * 16;
  Frag16 f;
  f.q[0] = *(const uint4*)(p);
  f.q[1] = *(const uint4*)(p + 8);
  return f.v;
}

// ---- LDS fragment loaders (GEMM kernels); tile is rows x 32, pitch LDSP ----
__device__ __forceinline__ v16bf lds_afrag(const unsigned short* sm, int row0) {
  const int lane = threadIdx.x & 31;
  const int r = lane & 15, c = lane >> 4;
  const unsigned short* p = sm + (row0 + r) * LDSP + c * 16;
  Frag16 f;
  f.q[0] = *(const uint4*)(p);
  f.q[1] = *(const uint4*)(p + 8);
  return f.v;
}

__device__ __forceinline__ v16bf lds_bfrag(const unsigned short* sm, int row0) {
  const int lane = threadIdx.x & 31;
  const int n = lane & 15, c = lane >> 4;
  const unsigned short* p = sm + (row0 + n) * LDSP + c * 16;
  Frag16 f;
  f.q[0] = *(const uint4*)(p);
  f.q[1] = *(const uint4*)(p + 8);
  return f.v;
}

__device__ __forceinline__ v8f wmma_bf16(v16bf a, v16bf b, v8f c) {
  return __builtin_amdgcn_wmma_f32_16x16x32_bf16(false, a, false, b, (short)0, c, false, false);
}

// CDNA5 async global->LDS copy, tracked by ASYNCcnt (ISA 08_async_tensor §4).
__device__ __forceinline__ void async_load16(unsigned lds_off, const unsigned short* gptr) {
  asm volatile("global_load_async_to_lds_b128 %0, %1, off"
               :: "v"(lds_off), "v"(gptr) : "memory");
}
__device__ __forceinline__ void wait_async0() {
  asm volatile("s_wait_asynccnt 0x0" ::: "memory");
}

// Stage a 128-row x 32-col bf16 k-slice into LDS (pitch LDSP); 256 threads x 2 b128 each.
__device__ __forceinline__ void fill_tile(unsigned short* sm, const unsigned short* g,
                                          int ld, int row0, int k0) {
  const int tid = threadIdx.x;
#pragma unroll
  for (int rep = 0; rep < 2; ++rep) {
    const int idx = tid + rep * 256;
    const int row = idx >> 2, cch = idx & 3;       // 4 x 16B chunks per 64B row
    const unsigned lds_off = (unsigned)(uintptr_t)(sm + row * LDSP + cch * 8);
    const unsigned short* gp = g + (size_t)(row0 + row) * ld + k0 + cch * 8;
    async_load16(lds_off, gp);
  }
}

__global__ __launch_bounds__(256)
void cvt_f32_bf16(const float* __restrict__ in, unsigned short* __restrict__ out, int n) {
  int i = (blockIdx.x * blockDim.x + threadIdx.x) * 4;
  if (i + 3 < n) {
    float4 f = *(const float4*)(in + i);
    union { unsigned short us[4]; uint2 u; } p;
    p.us[0] = f2bf(f.x); p.us[1] = f2bf(f.y); p.us[2] = f2bf(f.z); p.us[3] = f2bf(f.w);
    *(uint2*)(out + i) = p.u;
  }
}

// Shared GEMM pipeline: block tile 128M x 128N, 8 waves as (4M x 2N), wave tile 32M x 64N.
// Double-buffered async LDS staging of both A and B k-slices.
#define GEMM_CORE(Asrc, Wsrc, ACC)                                               \
  __shared__ __align__(16) unsigned short smA[2][128 * LDSP];                    \
  __shared__ __align__(16) unsigned short smB[2][128 * LDSP];                    \
  const int w   = threadIdx.x >> 5;                                              \
  const int mw0 = (w & 3) * 32;                                                  \
  const int nw0 = (w >> 2) * 64;                                                 \
  const int mb0 = blockIdx.x * 128;                                              \
  const int nb0 = blockIdx.y * 128;                                              \
  v8f ACC[8];                                                                    \
  _Pragma("unroll")                                                              \
  for (int i = 0; i < 8; ++i) ACC[i] = (v8f){0,0,0,0,0,0,0,0};                   \
  fill_tile(smA[0], Asrc, EMBED, mb0, 0);                                        \
  fill_tile(smB[0], Wsrc, EMBED, nb0, 0);                                        \
  wait_async0();                                                                 \
  __syncthreads();                                                               \
  for (int k0 = 0; k0 < EMBED; k0 += 32) {                                       \
    const int cur = (k0 >> 5) & 1;                                               \
    if (k0 + 32 < EMBED) {                                                       \
      fill_tile(smA[cur ^ 1], Asrc, EMBED, mb0, k0 + 32);                        \
      fill_tile(smB[cur ^ 1], Wsrc, EMBED, nb0, k0 + 32);                        \
    }                                                                            \
    v16bf bf[4];                                                                 \
    _Pragma("unroll")                                                            \
    for (int j = 0; j < 4; ++j) bf[j] = lds_bfrag(smB[cur], nw0 + 16 * j);       \
    _Pragma("unroll")                                                            \
    for (int a = 0; a < 2; ++a) {                                                \
      v16bf af = lds_afrag(smA[cur], mw0 + 16 * a);                              \
      _Pragma("unroll")                                                          \
      for (int j = 0; j < 4; ++j)                                                \
        ACC[a * 4 + j] = wmma_bf16(af, bf[j], ACC[a * 4 + j]);                   \
    }                                                                            \
    wait_async0();                                                               \
    __syncthreads();                                                             \
  }

// Fused QKV projection: y = x * W^T (+bias). blockIdx.z: 0=Q, 1=K, 2=V.
__global__ __launch_bounds__(256)
void qkv_gemm(const unsigned short* __restrict__ xb,
              const unsigned short* __restrict__ Wqb,
              const unsigned short* __restrict__ Wkb,
              const unsigned short* __restrict__ Wvb,
              const float* __restrict__ bq, const float* __restrict__ bk,
              const float* __restrict__ bv,
              unsigned short* __restrict__ Qb, unsigned short* __restrict__ Kb,
              unsigned short* __restrict__ Vtb,
              float* __restrict__ KhOut, float* __restrict__ VhOut) {
  const int mode = blockIdx.z;
  const unsigned short* W = (mode == 0) ? Wqb : (mode == 1) ? Wkb : Wvb;
  const float* bias = (mode == 0) ? bq : (mode == 1) ? bk : bv;

  GEMM_CORE(xb, W, acc)

  const int lane = threadIdx.x & 31;
  const int ln = lane & 15, cc = lane >> 4;
#pragma unroll
  for (int a = 0; a < 2; ++a) {
    const int m0 = mb0 + mw0 + a * 16;
#pragma unroll
    for (int j = 0; j < 4; ++j) {
      const int n = nb0 + nw0 + 16 * j + ln;
      const float bias_n = bias[n];
      const int h = n >> 7, d = n & (DH - 1);
      if (mode == 2) {
        union { unsigned short us[8]; uint4 u; } pk;
#pragma unroll
        for (int v = 0; v < 8; ++v) {
          const int m = m0 + v + cc * 8;
          const int bI = m >> 11, t = m & (TT - 1);
          const float val = acc[a * 4 + j][v] + bias_n;
          VhOut[((size_t)(bI * HEADS + h) * TT + t) * DH + d] = val;
          pk.us[v] = f2bf(val);
        }
        const int m = m0 + cc * 8;
        const int bI = m >> 11, t = m & (TT - 1);
        *(uint4*)(Vtb + ((size_t)(bI * HEADS + h) * DH + d) * TT + t) = pk.u;
      } else {
#pragma unroll
        for (int v = 0; v < 8; ++v) {
          const int m = m0 + v + cc * 8;
          const int bI = m >> 11, t = m & (TT - 1);
          const float val = acc[a * 4 + j][v] + bias_n;
          const size_t hs = ((size_t)(bI * HEADS + h) * TT + t) * DH + d;
          if (mode == 0) {
            Qb[hs] = f2bf(val);
          } else {
            Kb[hs] = f2bf(val);
            KhOut[hs] = val;
          }
        }
      }
    }
  }
}

// Flash attention per (b,h): one wave per 16-query tile.
// S^T = K_tile(16x128) x Q^T(128x16)  (keys = rows, queries = columns/lanes)
// O^T(128x16) += V^T(128x32) x P^T(32x16), with V stored pre-transposed [Dh][T].
__global__ __launch_bounds__(256)
void attn_kernel(const unsigned short* __restrict__ Qb,
                 const unsigned short* __restrict__ Kb,
                 const unsigned short* __restrict__ Vtb,
                 unsigned short* __restrict__ Ab) {
  const int bh = blockIdx.y;                 // 0..31
  const int b = bh >> 4, h = bh & 15;
  const int w = threadIdx.x >> 5;
  const int t0 = (blockIdx.x * 8 + w) * 16;  // query tile base
  const unsigned short* Qh = Qb  + (size_t)bh * TT * DH;
  const unsigned short* Kh = Kb  + (size_t)bh * TT * DH;
  const unsigned short* Vt = Vtb + (size_t)bh * DH * TT;

  const int lane = threadIdx.x & 31;
  const int ln = lane & 15, cc = lane >> 4;
  const int q = t0 + ln;                     // this lane's query index

  v16bf Bq[4];
#pragma unroll
  for (int kk = 0; kk < 4; ++kk) Bq[kk] = load_bfrag(Qh, DH, t0, kk * 32);

  v8f O[8];
#pragma unroll
  for (int i = 0; i < 8; ++i) O[i] = (v8f){0,0,0,0,0,0,0,0};
  float mrow = -3.0e38f, lrow = 0.0f;
  const float scale = 0.08838834764831845f;  // 1/sqrt(128)

  for (int j0 = 0; j0 <= t0 + 15; j0 += 32) {
    v8f s0 = (v8f){0,0,0,0,0,0,0,0};
    v8f s1 = (v8f){0,0,0,0,0,0,0,0};
#pragma unroll
    for (int kk = 0; kk < 4; ++kk) {
      s0 = wmma_bf16(load_afrag(Kh, DH, j0,      kk * 32), Bq[kk], s0);
      s1 = wmma_bf16(load_afrag(Kh, DH, j0 + 16, kk * 32), Bq[kk], s1);
    }
    // scale + causal mask; this lane holds keys j0+cc*8+v (s0) and +16 (s1)
    float p0[8], p1[8], mx = -3.0e38f;
#pragma unroll
    for (int v = 0; v < 8; ++v) {
      const int ki = j0 + v + cc * 8;
      float a0 = s0[v] * scale;
      float a1 = s1[v] * scale;
      if (ki > q)      a0 = -3.0e38f;
      if (ki + 16 > q) a1 = -3.0e38f;
      p0[v] = a0; p1[v] = a1;
      mx = fmaxf(mx, fmaxf(a0, a1));
    }
    mx = fmaxf(mx, __shfl_xor(mx, 16, 32));
    const float mnew = fmaxf(mrow, mx);
    const float alpha = __expf(mrow - mnew);
    float sum = 0.0f;
#pragma unroll
    for (int v = 0; v < 8; ++v) {
      p0[v] = __expf(p0[v] - mnew);
      p1[v] = __expf(p1[v] - mnew);
      sum += p0[v] + p1[v];
    }
    sum += __shfl_xor(sum, 16, 32);
    lrow = lrow * alpha + sum;
    mrow = mnew;

    // Repack P^T (32 keys x 16 queries) into a B fragment:
    // lane half cc needs keys cc*16 .. cc*16+15 of its query.
    Frag16 Pf;
#pragma unroll
    for (int v = 0; v < 8; ++v) {
      const float o0 = __shfl_xor(p0[v], 16, 32);
      const float o1 = __shfl_xor(p1[v], 16, 32);
      Pf.us[v]     = f2bf(cc == 0 ? p0[v] : o1);
      Pf.us[v + 8] = f2bf(cc == 0 ? o0    : p1[v]);
    }
    // O^T rescale + accumulate V^T x P^T  (A-frags contiguous from Vtb)
#pragma unroll
    for (int i = 0; i < 8; ++i) {
#pragma unroll
      for (int v = 0; v < 8; ++v) O[i][v] *= alpha;
      O[i] = wmma_bf16(load_afrag(Vt, TT, i * 16, j0), Pf.v, O[i]);
    }
  }

  // Normalize and store merged-head bf16 row: Ab[b*T + t][h*128 + d]
  const float inv = 1.0f / lrow;
  const size_t rowoff = ((size_t)(b * TT) + t0 + ln) * EMBED + h * DH;
#pragma unroll
  for (int i = 0; i < 8; ++i) {
    union { unsigned short us[8]; uint4 u; } pk;
#pragma unroll
    for (int v = 0; v < 8; ++v) pk.us[v] = f2bf(O[i][v] * inv);
    *(uint4*)(Ab + rowoff + i * 16 + cc * 8) = pk.u;
  }
}

// Output projection: out = Ab * Wo^T + bo  (f32 store)
__global__ __launch_bounds__(256)
void out_proj(const unsigned short* __restrict__ Ab,
              const unsigned short* __restrict__ Wob,
              const float* __restrict__ bo, float* __restrict__ out) {
  GEMM_CORE(Ab, Wob, acc)

  const int lane = threadIdx.x & 31;
  const int ln = lane & 15, cc = lane >> 4;
#pragma unroll
  for (int a = 0; a < 2; ++a) {
    const int m0 = mb0 + mw0 + a * 16;
#pragma unroll
    for (int j = 0; j < 4; ++j) {
      const int n = nb0 + nw0 + 16 * j + ln;
      const float bias_n = bo[n];
#pragma unroll
      for (int v = 0; v < 8; ++v) {
        const int m = m0 + v + cc * 8;
        out[(size_t)m * EMBED + n] = acc[a * 4 + j][v] + bias_n;
      }
    }
  }
}

extern "C" void kernel_launch(void* const* d_in, const int* in_sizes, int n_in,
                              void* d_out, int out_size, void* d_ws, size_t ws_size,
                              hipStream_t stream) {
  const float* x  = (const float*)d_in[0];
  const float* Wq = (const float*)d_in[1];
  const float* bq = (const float*)d_in[2];
  const float* Wk = (const float*)d_in[3];
  const float* bk = (const float*)d_in[4];
  const float* Wv = (const float*)d_in[5];
  const float* bv = (const float*)d_in[6];
  const float* Wo = (const float*)d_in[7];
  const float* bo = (const float*)d_in[8];

  const size_t NE = (size_t)NTOK * EMBED;   // 8,388,608 (== B*H*T*Dh)
  const size_t WE = (size_t)EMBED * EMBED;  // 4,194,304

  float* out   = (float*)d_out;
  float* KhOut = out + NE;
  float* VhOut = KhOut + NE;

  char* wsp = (char*)d_ws;
  size_t off = 0;
  auto take = [&](size_t bytes) {
    char* p = wsp + off;
    off = (off + bytes + 255) & ~(size_t)255;
    return p;
  };
  unsigned short* xb  = (unsigned short*)take(NE * 2);
  unsigned short* Wqb = (unsigned short*)take(WE * 2);
  unsigned short* Wkb = (unsigned short*)take(WE * 2);
  unsigned short* Wvb = (unsigned short*)take(WE * 2);
  unsigned short* Wob = (unsigned short*)take(WE * 2);
  unsigned short* Qb  = (unsigned short*)take(NE * 2);
  unsigned short* Kb  = (unsigned short*)take(NE * 2);
  unsigned short* Vtb = (unsigned short*)take(NE * 2);
  unsigned short* Ab  = (unsigned short*)take(NE * 2);
  (void)ws_size; (void)in_sizes; (void)n_in; (void)out_size;

  cvt_f32_bf16<<<(int)(NE / 4 / 256), 256, 0, stream>>>(x,  xb,  (int)NE);
  cvt_f32_bf16<<<(int)(WE / 4 / 256), 256, 0, stream>>>(Wq, Wqb, (int)WE);
  cvt_f32_bf16<<<(int)(WE / 4 / 256), 256, 0, stream>>>(Wk, Wkb, (int)WE);
  cvt_f32_bf16<<<(int)(WE / 4 / 256), 256, 0, stream>>>(Wv, Wvb, (int)WE);
  cvt_f32_bf16<<<(int)(WE / 4 / 256), 256, 0, stream>>>(Wo, Wob, (int)WE);

  qkv_gemm<<<dim3(NTOK / 128, EMBED / 128, 3), 256, 0, stream>>>(
      xb, Wqb, Wkb, Wvb, bq, bk, bv, Qb, Kb, Vtb, KhOut, VhOut);

  attn_kernel<<<dim3(TT / 16 / 8, BB * HEADS), 256, 0, stream>>>(Qb, Kb, Vtb, Ab);

  out_proj<<<dim3(NTOK / 128, EMBED / 128), 256, 0, stream>>>(Ab, Wob, bo, out);
}
